// MPN_27925877359025
// MI455X (gfx1250) — matrix operand order repaired and checked by reading, not compiled, not run on hardware
//
#include <hip/hip_runtime.h>
#include <hip/hip_bf16.h>
#include <cstdint>

// ---------------------------------------------------------------------------
// Problem constants (match reference)
// ---------------------------------------------------------------------------
#define NM    8192
#define NA    98304
#define NB    196608
#define NAp1  (NA + 1)
#define NBp1  (NB + 1)
#define HID   300
#define HEADS 4
#define DEPTH 6
#define MD    600
#define AF    133
#define BF    147

// zero-padded K strides (multiples of 64 -> aligned b128 frags + 64-wide K chunks)
#define KP_BF   192   // 147 -> 192   (fbonds / W_i)
#define KP_HID  320   // 300 -> 320   (message / W_ma / per_bond / W_mi)
#define KP_MD   640   // 600 -> 640   (master / W_mo)
#define KP_4H  1216   // 1200 -> 1216 (comps / W_h)
#define KP_AIN  448   // 433 -> 448   (atom input / W_o)

#define LDS_ROW 72    // bf16 elements per LDS B row (64 data + 8 pad; 144B, 16B-aligned)

typedef __bf16 bf16_t;
typedef __attribute__((ext_vector_type(16))) __bf16 v16bf;
typedef __attribute__((ext_vector_type(8)))  __bf16 bf16x8;
typedef __attribute__((ext_vector_type(8)))  float  v8f;

// two aligned 16B loads -> one 16-element bf16 fragment
__device__ __forceinline__ v16bf ld_frag2(const bf16_t* p0, const bf16_t* p1) {
  union U { v16bf v; bf16x8 h[2]; } u;
  u.h[0] = *(const bf16x8*)p0;
  u.h[1] = *(const bf16x8*)p1;
  return u.v;
}

// ---------------------------------------------------------------------------
// Generic WMMA GEMM on pre-packed bf16 operands (zero-padded K = KPAD):
//   C[M,N] = epilogue(A[M,KPAD] @ W[N,KPAD]^T)
// block = 256 = 8 waves; wave computes 32(M) x 64(N); block tile 256(M) x 64(N).
// B panel is staged through LDS in 64-K chunks with global_load_async_to_lds,
// double-buffered; A fragments are register double-buffered one chunk ahead.
// Fragment layouts per CDNA5 ISA 7.12.2 (wave32):
//   A: M=lane%16; half h: elems 0..7 -> K=kb+8h.., elems 8..15 -> K=kb+16+8h..
//   B: N=lane%16; half h: elems 0..7 -> K=kb+8h.., elems 8..15 -> K=kb+16+8h..
//   C/D: 8 VGPRs; M = r + 8*(lane>=16), N = lane%16
// ---------------------------------------------------------------------------
enum { EPI_PLAIN = 0, EPI_RELU_F32 = 1, EPI_RELU_BF16 = 2, EPI_DUAL = 3, EPI_FUSED = 4 };

template <int MODE, int KPAD>
__global__ __launch_bounds__(256) void gemm_wmma(
    const bf16_t* __restrict__ A,
    const bf16_t* __restrict__ W,
    float* __restrict__ C, long ldc,
    float* __restrict__ C2,                 // DUAL: relu f32 copy
    bf16_t* __restrict__ Cb, long ldcb,     // bf16 output copy
    const float* __restrict__ bias,
    const float* __restrict__ add1,
    const float* __restrict__ add2,
    int M, int N)
{
  constexpr int NCHUNK = KPAD / 64;
  __shared__ bf16_t smem[2][64 * LDS_ROW];

  const int lane  = threadIdx.x & 31;
  const int wave  = threadIdx.x >> 5;
  const int half  = lane >> 4;
  const int mbase = blockIdx.x * 256 + wave * 32;
  const int nbase = blockIdx.y * 64;

  int arow0 = mbase + (lane & 15);      if (arow0 >= M) arow0 = M - 1;
  int arow1 = mbase + 16 + (lane & 15); if (arow1 >= M) arow1 = M - 1;
  const bf16_t* ap0 = A + (long)arow0 * KPAD + half * 8;
  const bf16_t* ap1 = A + (long)arow1 * KPAD + half * 8;

  // async-stage one 64x64 B chunk into LDS (whole block: 512 x 16B segments)
  auto stage = [&](int c) {
    const int kb = c * 64;
    const int t  = threadIdx.x;
#pragma unroll
    for (int rep = 0; rep < 2; ++rep) {
      int s  = t + rep * 256;
      int n  = s >> 3;
      int k8 = (s & 7) * 8;
      int nrow = nbase + n; if (nrow >= N) nrow = N - 1;
      const bf16_t* g = W + (long)nrow * KPAD + kb + k8;
      uint32_t l = (uint32_t)(uintptr_t)&smem[c & 1][n * LDS_ROW + k8];
      asm volatile("global_load_async_to_lds_b128 %0, %1, off"
                   :: "v"(l), "v"(g) : "memory");
    }
  };
  auto loadA4 = [&](int kc, v16bf Af[4]) {
    Af[0] = ld_frag2(ap0 + kc,      ap0 + kc + 16);   // row set 0, K-step 0
    Af[1] = ld_frag2(ap1 + kc,      ap1 + kc + 16);   // row set 1, K-step 0
    Af[2] = ld_frag2(ap0 + kc + 32, ap0 + kc + 48);   // row set 0, K-step 1
    Af[3] = ld_frag2(ap1 + kc + 32, ap1 + kc + 48);   // row set 1, K-step 1
  };

  v8f acc[2][4] = {};
  v16bf Ac[4];
  stage(0);
  loadA4(0, Ac);
  asm volatile("s_wait_asynccnt 0x0" ::: "memory");
  __syncthreads();

  for (int c = 0; c < NCHUNK; ++c) {
    v16bf An[4];
    const bool more = (c + 1 < NCHUNK);
    if (more) {
      stage(c + 1);                 // async B prefetch -> other LDS buffer
      loadA4((c + 1) * 64, An);     // A prefetch -> registers
    }
    const bf16_t* sb = &smem[c & 1][0];
#pragma unroll
    for (int j = 0; j < 4; ++j) {
      const bf16_t* bp = sb + (j * 16 + (lane & 15)) * LDS_ROW + half * 8;
      v16bf b0 = ld_frag2(bp, bp + 16);               // K-step 0 (ds_load_b128 x2)
      acc[0][j] = __builtin_amdgcn_wmma_f32_16x16x32_bf16(
          false, Ac[0], false, b0, (short)0, acc[0][j], false, false);
      acc[1][j] = __builtin_amdgcn_wmma_f32_16x16x32_bf16(
          false, Ac[1], false, b0, (short)0, acc[1][j], false, false);
      v16bf b1 = ld_frag2(bp + 32, bp + 48);          // K-step 1
      acc[0][j] = __builtin_amdgcn_wmma_f32_16x16x32_bf16(
          false, Ac[2], false, b1, (short)0, acc[0][j], false, false);
      acc[1][j] = __builtin_amdgcn_wmma_f32_16x16x32_bf16(
          false, Ac[3], false, b1, (short)0, acc[1][j], false, false);
    }
    if (more) {
      asm volatile("s_wait_asynccnt 0x0" ::: "memory");
    }
    __syncthreads();
    if (more) {
      Ac[0] = An[0]; Ac[1] = An[1]; Ac[2] = An[2]; Ac[3] = An[3];
    }
  }

#pragma unroll
  for (int p = 0; p < 2; ++p) {
    const int mrow0 = mbase + p * 16 + half * 8;
#pragma unroll
    for (int j = 0; j < 4; ++j) {
      int ncol = nbase + j * 16 + (lane & 15);
      if (ncol >= N) continue;
      float bval = 0.0f;
      if (MODE == EPI_RELU_F32 || MODE == EPI_RELU_BF16 || MODE == EPI_FUSED)
        bval = bias[ncol];
#pragma unroll
      for (int r = 0; r < 8; ++r) {
        int m = mrow0 + r;
        if (m >= M) continue;
        float v = acc[p][j][r];
        if (MODE == EPI_PLAIN) {
          C[(long)m * ldc + ncol] = v;
        } else if (MODE == EPI_RELU_F32) {
          C[(long)m * ldc + ncol] = fmaxf(v + bval, 0.0f);
        } else if (MODE == EPI_RELU_BF16) {
          Cb[(long)m * ldcb + ncol] = (bf16_t)fmaxf(v + bval, 0.0f);
        } else if (MODE == EPI_DUAL) {
          long ix = (long)m * ldc + ncol;
          C[ix] = v;
          float rr = fmaxf(v, 0.0f);
          C2[ix] = rr;
          Cb[(long)m * ldcb + ncol] = (bf16_t)rr;
        } else {                                     // EPI_FUSED
          long ix = (long)m * ldc + ncol;
          float s = fmaxf(v + bval + add1[ix] + add2[ix], 0.0f);
          C[ix] = s;
          Cb[(long)m * ldcb + ncol] = (bf16_t)s;
        }
      }
    }
  }
}

// ---------------------------------------------------------------------------
// Fused attention: one wave per (bond, head).
//   scores[n] = q[b,h,:] . message[bgraph[b,n],:]; attn = softmax;
//   comps_bf[b,h,:] = sum_n attn[n]*nei[n,:]   (written directly as bf16)
// ---------------------------------------------------------------------------
__global__ __launch_bounds__(256) void attn_comps(
    const float* __restrict__ msg,      // [NBp1, HID] f32
    const float* __restrict__ q,        // [NBp1, HEADS*HID] f32
    bf16_t* __restrict__ comps,         // [NBp1, KP_4H] bf16 (pads stay zero)
    const int* __restrict__ bgraph)     // [NBp1, 6]
{
  const long wid = (long)blockIdx.x * 8 + (threadIdx.x >> 5);
  if (wid >= (long)NBp1 * HEADS) return;
  const int b = (int)(wid >> 2);
  const int h = (int)(wid & 3);
  const int lane = threadIdx.x & 31;

  const float* qp = q + (long)b * (HEADS * HID) + (long)h * HID;
  float qreg[10];
#pragma unroll
  for (int i = 0; i < 10; ++i) {
    int d = lane + 32 * i;
    qreg[i] = (d < HID) ? qp[d] : 0.0f;
  }

  int nb[6];
#pragma unroll
  for (int n = 0; n < 6; ++n) nb[n] = bgraph[(long)b * 6 + n];

  float nreg[6][10];
#pragma unroll
  for (int n = 0; n < 6; ++n) {
    const float* mp = msg + (long)nb[n] * HID;
#pragma unroll
    for (int i = 0; i < 10; ++i) {
      int d = lane + 32 * i;
      nreg[n][i] = (d < HID) ? mp[d] : 0.0f;
    }
  }

  float sc[6];
#pragma unroll
  for (int n = 0; n < 6; ++n) {
    float s = 0.0f;
#pragma unroll
    for (int i = 0; i < 10; ++i) s += qreg[i] * nreg[n][i];
#pragma unroll
    for (int off = 16; off > 0; off >>= 1) s += __shfl_xor(s, off, 32);
    sc[n] = s;
  }

  float mx = sc[0];
#pragma unroll
  for (int n = 1; n < 6; ++n) mx = fmaxf(mx, sc[n]);
  float ex[6], sum = 0.0f;
#pragma unroll
  for (int n = 0; n < 6; ++n) { ex[n] = __expf(sc[n] - mx); sum += ex[n]; }
  const float inv = 1.0f / sum;

  bf16_t* cp = comps + (long)b * KP_4H + (long)h * HID;
#pragma unroll
  for (int i = 0; i < 10; ++i) {
    float c = 0.0f;
#pragma unroll
    for (int n = 0; n < 6; ++n) c += (ex[n] * inv) * nreg[n][i];
    int d = lane + 32 * i;
    if (d < HID) cp[d] = (bf16_t)c;
  }
}

// ---------------------------------------------------------------------------
// Helpers
// ---------------------------------------------------------------------------
__global__ void zero_f32(float* __restrict__ p, long n) {
  long i = (long)blockIdx.x * blockDim.x + threadIdx.x;
  if (i < n) p[i] = 0.0f;
}

__global__ void count_ids(const int* __restrict__ ids, int n, float* __restrict__ cnt) {
  int i = blockIdx.x * blockDim.x + threadIdx.x;
  if (i < n) atomicAdd(&cnt[ids[i]], 1.0f);
}

// pack f32 [rows,K] -> bf16 [rows,ldd] with zero pad
__global__ void pack_bf16(const float* __restrict__ src, bf16_t* __restrict__ dst,
                          long rows, int K, int ldd) {
  long i = (long)blockIdx.x * blockDim.x + threadIdx.x;
  if (i >= rows * (long)ldd) return;
  long r = i / ldd;
  int  c = (int)(i % ldd);
  dst[i] = (c < K) ? (bf16_t)src[r * K + c] : (bf16_t)0.0f;
}

// W_ma [h][e][d] f32 -> wt [(h*HID+d)][KP_HID] bf16 at col e  (i.e. W_ma^T per head)
__global__ void pack_wmaT(const float* __restrict__ w, bf16_t* __restrict__ wt) {
  long i = (long)blockIdx.x * blockDim.x + threadIdx.x;
  if (i >= (long)HEADS * HID * KP_HID) return;
  long row = i / KP_HID;
  int  e   = (int)(i % KP_HID);
  int  h   = (int)(row / HID);
  int  d   = (int)(row % HID);
  wt[i] = (e < HID) ? (bf16_t)w[((long)h * HID + e) * HID + d] : (bf16_t)0.0f;
}

// dst[ids[r],:] += src[r+1,:]   (skip padding row 0)
__global__ void scatter_add_rows(const float* __restrict__ src,
                                 const int* __restrict__ ids,
                                 float* __restrict__ dst) {
  int r = blockIdx.x;
  int d = threadIdx.x;
  if (d >= HID) return;
  atomicAdd(&dst[(long)ids[r] * HID + d], src[(long)(r + 1) * HID + d]);
}

// per_bond[r,:] = (r==0) ? 0 : mol_mean[ids[r-1],:] / max(cnt,1)  -> bf16, stride KP_HID
__global__ void gather_master_in(const float* __restrict__ mol,
                                 const float* __restrict__ cnt,
                                 const int* __restrict__ ids,
                                 bf16_t* __restrict__ out) {
  int r = blockIdx.x;
  int d = threadIdx.x;
  if (d >= HID) return;
  float v = 0.0f;
  if (r > 0) {
    int m = ids[r - 1];
    v = mol[(long)m * HID + d] / fmaxf(cnt[m], 1.0f);
  }
  out[(long)r * KP_HID + d] = (bf16_t)v;
}

// ainput[a,0:133]=fatoms ; [133:433]=sum_n message[agraph[a,n]] ; [433:448]=0 (bf16)
__global__ __launch_bounds__(448) void atom_pack(
    const float* __restrict__ fatoms, const float* __restrict__ msg,
    const int* __restrict__ agraph, bf16_t* __restrict__ out) {
  int a = blockIdx.x;
  int t = threadIdx.x;
  float v = 0.0f;
  if (t < AF) {
    v = fatoms[(long)a * AF + t];
  } else if (t < AF + HID) {
    int d = t - AF;
    float s = 0.0f;
#pragma unroll
    for (int n = 0; n < 6; ++n)
      s += msg[(long)agraph[(long)a * 6 + n] * HID + d];
    v = s;
  }
  out[(long)a * KP_AIN + t] = (bf16_t)v;
}

__global__ void divide_rows(float* __restrict__ dst, const float* __restrict__ cnt, long n) {
  long i = (long)blockIdx.x * blockDim.x + threadIdx.x;
  if (i < n) {
    int m = (int)(i / HID);
    dst[i] /= fmaxf(cnt[m], 1.0f);
  }
}

// ---------------------------------------------------------------------------
// Launch
// ---------------------------------------------------------------------------
extern "C" void kernel_launch(void* const* d_in, const int* in_sizes, int n_in,
                              void* d_out, int out_size, void* d_ws, size_t ws_size,
                              hipStream_t stream) {
  (void)in_sizes; (void)n_in; (void)out_size; (void)ws_size;
  const float* fatoms   = (const float*)d_in[0];
  const float* fbonds   = (const float*)d_in[1];
  const int*   agraph   = (const int*)d_in[2];
  const int*   bgraph   = (const int*)d_in[3];
  const int*   atom2mol = (const int*)d_in[4];
  const int*   bond2mol = (const int*)d_in[5];
  const float* W_i      = (const float*)d_in[6];
  const float* W_h      = (const float*)d_in[7];
  const float* W_ma     = (const float*)d_in[8];
  const float* W_mi     = (const float*)d_in[9];
  const float* b_mi     = (const float*)d_in[10];
  const float* W_mo     = (const float*)d_in[11];
  const float* b_mo     = (const float*)d_in[12];
  const float* W_o      = (const float*)d_in[13];
  const float* b_o      = (const float*)d_in[14];
  float* out = (float*)d_out;

  // ---- workspace layout: f32 region, then contiguous bf16 region -----------
  float* ws = (float*)d_ws;
  long off = 0;
  float* binput   = ws + off; off += (long)NBp1 * HID;
  float* message  = ws + off; off += (long)NBp1 * HID;
  float* qbuf     = ws + off; off += (long)NBp1 * HEADS * HID;   // q (f32)
  float* nei_msg  = ws + off; off += (long)NBp1 * HID;           // also atom_h
  float* mol_mean = ws + off; off += (long)NM * HID;
  float* bond_cnt = ws + off; off += NM;
  float* atom_cnt = ws + off; off += NM;

  bf16_t* bfbase = (bf16_t*)(ws + off);
  long boff = 0;
  bf16_t* fbonds_bf  = bfbase + boff; boff += (long)NBp1 * KP_BF;
  bf16_t* message_bf = bfbase + boff; boff += (long)NBp1 * KP_HID;  // also per_bond
  bf16_t* comps_bf   = bfbase + boff; boff += (long)NBp1 * KP_4H;   // also ainput (stride 448)
  bf16_t* master_bf  = bfbase + boff; boff += (long)NBp1 * KP_MD;
  bf16_t* Wi_bf      = bfbase + boff; boff += (long)HID * KP_BF;
  bf16_t* WmaT_bf    = bfbase + boff; boff += (long)HEADS * HID * KP_HID;
  bf16_t* Wh_bf      = bfbase + boff; boff += (long)HID * KP_4H;
  bf16_t* Wmi_bf     = bfbase + boff; boff += (long)MD * KP_HID;
  bf16_t* Wmo_bf     = bfbase + boff; boff += (long)HID * KP_MD;
  bf16_t* Wo_bf      = bfbase + boff; boff += (long)HID * KP_AIN;
  const long bf_total = boff;   // even

  auto ggrid = [](int M, int N) { return dim3((M + 255) / 256, (N + 63) / 64); };
  auto lgrid = [](long n) { return (unsigned)((n + 255) / 256); };

  // ---- one-time init: zero bf16 region (pads stay zero), counts, packs -----
  zero_f32<<<lgrid(bf_total / 2), 256, 0, stream>>>((float*)bfbase, bf_total / 2);
  zero_f32<<<lgrid(2L * NM), 256, 0, stream>>>(bond_cnt, 2L * NM);  // bond+atom cnt contiguous
  count_ids<<<lgrid(NB), 256, 0, stream>>>(bond2mol, NB, bond_cnt);
  count_ids<<<lgrid(NA), 256, 0, stream>>>(atom2mol, NA, atom_cnt);

  pack_bf16<<<lgrid((long)NBp1 * KP_BF), 256, 0, stream>>>(fbonds, fbonds_bf, NBp1, BF, KP_BF);
  pack_bf16<<<lgrid((long)HID * KP_BF), 256, 0, stream>>>(W_i, Wi_bf, HID, BF, KP_BF);
  pack_bf16<<<lgrid((long)HID * KP_4H), 256, 0, stream>>>(W_h, Wh_bf, HID, HEADS * HID, KP_4H);
  pack_bf16<<<lgrid((long)MD * KP_HID), 256, 0, stream>>>(W_mi, Wmi_bf, MD, HID, KP_HID);
  pack_bf16<<<lgrid((long)HID * KP_MD), 256, 0, stream>>>(W_mo, Wmo_bf, HID, MD, KP_MD);
  pack_bf16<<<lgrid((long)HID * KP_AIN), 256, 0, stream>>>(W_o, Wo_bf, HID, AF + HID, KP_AIN);
  pack_wmaT<<<lgrid((long)HEADS * HID * KP_HID), 256, 0, stream>>>(W_ma, WmaT_bf);

  // ---- binput = fbonds @ W_i^T ; message = relu(binput) (f32 + bf16) -------
  gemm_wmma<EPI_DUAL, KP_BF><<<ggrid(NBp1, HID), 256, 0, stream>>>(
      fbonds_bf, Wi_bf, binput, HID, message, message_bf, KP_HID,
      nullptr, nullptr, nullptr, NBp1, HID);

  for (int it = 0; it < DEPTH - 1; ++it) {
    // q[b,h,:] = message @ W_ma[h]
    for (int h = 0; h < HEADS; ++h)
      gemm_wmma<EPI_PLAIN, KP_HID><<<ggrid(NBp1, HID), 256, 0, stream>>>(
          message_bf, WmaT_bf + (long)h * HID * KP_HID,
          qbuf + (long)h * HID, (long)HEADS * HID, nullptr, nullptr, 0,
          nullptr, nullptr, nullptr, NBp1, HID);

    // scores -> softmax -> comps (bf16, padded stride)
    {
      long nwaves = (long)NBp1 * HEADS;
      attn_comps<<<(unsigned)((nwaves + 7) / 8), 256, 0, stream>>>(message, qbuf, comps_bf, bgraph);
    }

    // nei_msg = comps @ W_h^T  (K = 1200)
    gemm_wmma<EPI_PLAIN, KP_4H><<<ggrid(NBp1, HID), 256, 0, stream>>>(
        comps_bf, Wh_bf, nei_msg, HID, nullptr, nullptr, 0,
        nullptr, nullptr, nullptr, NBp1, HID);

    // mol_mean = segment_sum(nei_msg[1:]); per_bond = mol_mean[bond2mol]/cnt (bf16)
    zero_f32<<<lgrid((long)NM * HID), 256, 0, stream>>>(mol_mean, (long)NM * HID);
    scatter_add_rows<<<NB, 320, 0, stream>>>(nei_msg, bond2mol, mol_mean);
    gather_master_in<<<NBp1, 320, 0, stream>>>(mol_mean, bond_cnt, bond2mol, message_bf);

    // master = relu(per_bond @ W_mi^T + b_mi)  -> bf16 only
    gemm_wmma<EPI_RELU_BF16, KP_HID><<<ggrid(NBp1, MD), 256, 0, stream>>>(
        message_bf, Wmi_bf, nullptr, 0, nullptr, master_bf, KP_MD,
        b_mi, nullptr, nullptr, NBp1, MD);

    // message = relu(binput + nei_msg + master @ W_mo^T + b_mo)  (f32 + bf16)
    gemm_wmma<EPI_FUSED, KP_MD><<<ggrid(NBp1, HID), 256, 0, stream>>>(
        master_bf, Wmo_bf, message, HID, nullptr, message_bf, KP_HID,
        b_mo, binput, nei_msg, NBp1, HID);
  }

  // ---- atom readout --------------------------------------------------------
  atom_pack<<<NAp1, 448, 0, stream>>>(fatoms, message, agraph, comps_bf);

  gemm_wmma<EPI_RELU_F32, KP_AIN><<<ggrid(NAp1, HID), 256, 0, stream>>>(
      comps_bf, Wo_bf, nei_msg, HID, nullptr, nullptr, 0,
      b_o, nullptr, nullptr, NAp1, HID);

  zero_f32<<<lgrid((long)NM * HID), 256, 0, stream>>>(out, (long)NM * HID);
  scatter_add_rows<<<NA, 320, 0, stream>>>(nei_msg, atom2mol, out);
  divide_rows<<<lgrid((long)NM * HID), 256, 0, stream>>>(out, atom_cnt, (long)NM * HID);
}
